// MySimplicialConvolution_63677185130898
// MI455X (gfx1250) — matrix operand order repaired and checked by reading, not compiled, not run on hardware
//
#include <hip/hip_runtime.h>

// ---------------------------------------------------------------------------
// Chebyshev graph-conv for MI455X (gfx1250, wave32, WMMA).
//   T_k = 2*(T_{k-1} @ L^T) - T_{k-2};  y = einsum('imk,oik->om', X, theta)+b
// Strategy: L (268MB fp32) -> f16 once (134MB, fits 192MB L2); 4 passes over
// Lh with v_wmma_f32_16x16x32_f16, f32 accumulation; recurrence math in f32.
// ---------------------------------------------------------------------------

typedef _Float16 h8   __attribute__((ext_vector_type(8)));
typedef _Float16 h4   __attribute__((ext_vector_type(4)));
typedef _Float16 v16h __attribute__((ext_vector_type(16)));
typedef float    v8f  __attribute__((ext_vector_type(8)));
typedef float    f4   __attribute__((ext_vector_type(4)));

#define NN    8192
#define C_IN  32
#define C_OUT 64
#define KF    5

// ---- fp32 -> f16 down-convert (vectorized, n divisible by 4) --------------
__global__ __launch_bounds__(256) void cvt_f32_f16(const float* __restrict__ src,
                                                   _Float16* __restrict__ dst,
                                                   size_t n) {
    size_t i = ((size_t)blockIdx.x * blockDim.x + threadIdx.x) * 4;
    if (i + 3 < n) {
        f4 v = *(const f4*)(src + i);
        h4 o;
        o.x = (_Float16)v.x; o.y = (_Float16)v.y;
        o.z = (_Float16)v.z; o.w = (_Float16)v.w;
        *(h4*)(dst + i) = o;
    }
}

// ---- one Chebyshev step: Tn = alpha*(Tc @ Lh^T) - beta*Tp -----------------
// One wave per 16-column output tile; 2 f32 accumulators cover 32 channels.
// ISA 7.12.2 layouts (wave32, f16 16x16x32):
//   A: lane l<16 -> row m=l, K = {0..7}u{16..23}; lane>=16 -> K={8..15}u{24..31}
//   B: lane l    -> col n=l%16, K same as A: {8h..8h+7}u{16+8h..16+8h+7}, h=l/16
//   C/D: lane l  -> col n=l%16, VGPR v -> row m = v + 8*(l/16)
__global__ __launch_bounds__(64) void cheb_step(const _Float16* __restrict__ Lh,
                                                const _Float16* __restrict__ Tc,
                                                const float*    __restrict__ Tp,
                                                float*          __restrict__ Tn,
                                                _Float16*       __restrict__ Tnh,
                                                float alpha, float beta) {
    const int lane = threadIdx.x & 31;
    const int wave = threadIdx.x >> 5;
    const int j0   = (blockIdx.x * 2 + wave) * 16;   // 512 tiles total
    const int half = lane >> 4;
    const int nidx = lane & 15;

    // B stream: row (j0+nidx) of Lh, this lane covers K {8*half..+7}u{16+8*half..+7}
    const _Float16* bp  = Lh + (size_t)(j0 + nidx) * NN + 8 * half;
    // A streams: channel rows nidx and nidx+16 of Tc
    const _Float16* a0p = Tc + (size_t)nidx * NN;
    const _Float16* a1p = Tc + (size_t)(nidx + 16) * NN;
    const int aoff0 = 8 * half;        // K 0..7  (lanes<16) / 8..15  (lanes>=16)
    const int aoff1 = 16 + 8 * half;   // K 16..23(lanes<16) / 24..31 (lanes>=16)

    v8f acc0 = {};
    v8f acc1 = {};

    #pragma unroll 2
    for (int k0 = 0; k0 < NN; k0 += 32) {
        // prefetch the L stream ahead (matters on the first, HBM-fed pass)
        __builtin_prefetch(bp + k0 + 2048, 0, 0);

        union { v16h v; h8 h[2]; } A0, A1, B;
        A0.h[0] = *(const h8*)(a0p + k0 + aoff0);
        A0.h[1] = *(const h8*)(a0p + k0 + aoff1);
        A1.h[0] = *(const h8*)(a1p + k0 + aoff0);
        A1.h[1] = *(const h8*)(a1p + k0 + aoff1);
        B.h[0]  = *(const h8*)(bp + k0);
        B.h[1]  = *(const h8*)(bp + k0 + 16);

        acc0 = __builtin_amdgcn_wmma_f32_16x16x32_f16(false, A0.v, false, B.v,
                                                      (short)0, acc0, false, false);
        acc1 = __builtin_amdgcn_wmma_f32_16x16x32_f16(false, A1.v, false, B.v,
                                                      (short)0, acc1, false, false);
    }

    // Recurrence combine in f32; emit both f32 (for -T_prev / final einsum)
    // and f16 (next step's A operand).
    const size_t jc = (size_t)(j0 + nidx);
    #pragma unroll
    for (int v = 0; v < 8; ++v) {
        const int c0 = v + 8 * half;          // channels 0..15
        size_t i0 = (size_t)c0 * NN + jc;
        float r0 = alpha * acc0[v] - beta * Tp[i0];
        Tn[i0]  = r0;
        Tnh[i0] = (_Float16)r0;

        const int c1 = c0 + 16;               // channels 16..31
        size_t i1 = (size_t)c1 * NN + jc;
        float r1 = alpha * acc1[v] - beta * Tp[i1];
        Tn[i1]  = r1;
        Tnh[i1] = (_Float16)r1;
    }
}

// ---- final contraction: y[o,m] = sum_{i,k} T_k[i,m]*theta[o,i,k] + b[o] ---
__global__ __launch_bounds__(256) void assemble(const float* __restrict__ T0,
                                                const float* __restrict__ T1,
                                                const float* __restrict__ T2,
                                                const float* __restrict__ T3,
                                                const float* __restrict__ T4,
                                                const float* __restrict__ theta,
                                                const float* __restrict__ bias,
                                                float* __restrict__ y) {
    const int idx = blockIdx.x * blockDim.x + threadIdx.x;   // C_OUT*NN threads
    const int o = idx >> 13;
    const int m = idx & (NN - 1);
    const float* Ts[KF] = {T0, T1, T2, T3, T4};

    float acc = bias[o];
    #pragma unroll 4
    for (int i = 0; i < C_IN; ++i) {
        const float* th = theta + ((size_t)o * C_IN + i) * KF;  // scalar-uniform
        const size_t base = (size_t)i * NN + m;                 // coalesced
        #pragma unroll
        for (int k = 0; k < KF; ++k)
            acc += Ts[k][base] * th[k];
    }
    y[idx] = acc;
}

// ---------------------------------------------------------------------------
extern "C" void kernel_launch(void* const* d_in, const int* in_sizes, int n_in,
                              void* d_out, int out_size, void* d_ws, size_t ws_size,
                              hipStream_t stream) {
    const float* L     = (const float*)d_in[0];   // (N, N)
    const float* x     = (const float*)d_in[1];   // (C_IN, N)  == T_0 (f32)
    const float* theta = (const float*)d_in[2];   // (C_OUT, C_IN, 5)
    const float* bias  = (const float*)d_in[3];   // (C_OUT, 1)
    float* y = (float*)d_out;

    // Workspace layout:
    //   Lh  : N*N f16                             = 134,217,728 B
    //   Th  : 5 x (C_IN*N) f16 (T0..T4 operands)  =   2,621,440 B
    //   Tf  : 4 x (C_IN*N) f32 (T1..T4)           =   4,194,304 B
    char* ws = (char*)d_ws;
    _Float16* Lh = (_Float16*)ws;
    const size_t szLh = (size_t)NN * NN * sizeof(_Float16);
    const size_t szTh = (size_t)C_IN * NN * sizeof(_Float16);
    const size_t szTf = (size_t)C_IN * NN * sizeof(float);

    _Float16* Th[5];
    for (int k = 0; k < 5; ++k) Th[k] = (_Float16*)(ws + szLh + (size_t)k * szTh);
    float* Tf[5];
    Tf[0] = nullptr; // T_0 in f32 is the input x itself
    for (int k = 1; k < 5; ++k)
        Tf[k] = (float*)(ws + szLh + 5 * szTh + (size_t)(k - 1) * szTf);

    // 1) down-convert L and x to f16
    {
        size_t nL = (size_t)NN * NN;
        cvt_f32_f16<<<(unsigned)(nL / 4 / 256), 256, 0, stream>>>(L, Lh, nL);
        size_t nX = (size_t)C_IN * NN;
        cvt_f32_f16<<<(unsigned)(nX / 4 / 256), 256, 0, stream>>>(x, Th[0], nX);
    }

    // 2) Chebyshev recurrence: 4 WMMA passes over the (L2-resident) f16 L.
    //    T1 = T0 @ L^T              (alpha=1, beta=0; Tp read is harmless)
    //    Tk = 2*(T_{k-1}@L^T) - T_{k-2}
    cheb_step<<<256, 64, 0, stream>>>(Lh, Th[0], x,     Tf[1], Th[1], 1.0f, 0.0f);
    cheb_step<<<256, 64, 0, stream>>>(Lh, Th[1], x,     Tf[2], Th[2], 2.0f, 1.0f);
    cheb_step<<<256, 64, 0, stream>>>(Lh, Th[2], Tf[1], Tf[3], Th[3], 2.0f, 1.0f);
    cheb_step<<<256, 64, 0, stream>>>(Lh, Th[3], Tf[2], Tf[4], Th[4], 2.0f, 1.0f);

    // 3) channel mixing (f32 throughout)
    assemble<<<(C_OUT * NN) / 256, 256, 0, stream>>>(x, Tf[1], Tf[2], Tf[3], Tf[4],
                                                     theta, bias, y);
}
